// TransformerLayer_4879082848955
// MI455X (gfx1250) — matrix operand order repaired and checked by reading, not compiled, not run on hardware
//
#include <hip/hip_runtime.h>

// ---------------------------------------------------------------------------
// CDNA5 (gfx1250) transformer layer: bf16 WMMA GEMMs + flash attention.
// Wave32. WMMA shape 16x16x32 bf16 -> f32.
// Weights are pre-transposed at bf16-conversion time so ALL LDS staging is
// b128 loads/stores (no scalar scatter). Async global->LDS staging (ASYNCcnt).
// ---------------------------------------------------------------------------

typedef unsigned short u16;
typedef __bf16 v16bf __attribute__((ext_vector_type(16)));
typedef float  v8f   __attribute__((ext_vector_type(8)));
typedef unsigned int u32x4 __attribute__((ext_vector_type(4)));
typedef int i32x4 __attribute__((ext_vector_type(4)));

union BF16x16 { u32x4 q[2]; v16bf v; };

#if defined(__has_builtin)
#if __has_builtin(__builtin_amdgcn_global_load_async_to_lds_b128) && \
    __has_builtin(__builtin_amdgcn_s_wait_asynccnt)
#define HAVE_ASYNC_LDS 1
#endif
#endif

__device__ inline void stage_b128(const u16* gsrc, u16* ldst) {
#if defined(HAVE_ASYNC_LDS)
    __builtin_amdgcn_global_load_async_to_lds_b128(
        (__attribute__((address_space(1))) i32x4*)gsrc,
        (__attribute__((address_space(3))) i32x4*)ldst, 0, 0);
#else
    *(u32x4*)ldst = *(const u32x4*)gsrc;
#endif
}
__device__ inline void stage_wait() {
#if defined(HAVE_ASYNC_LDS)
    __builtin_amdgcn_s_wait_asynccnt(0);
#endif
}

__device__ inline u16 f2bf(float f) {
    unsigned int u = __float_as_uint(f);
    unsigned int r = (u + 0x7FFFu + ((u >> 16) & 1u)) >> 16;
    return (u16)r;
}

__device__ inline v8f wmma_bf16(const BF16x16& a, const BF16x16& b, v8f c) {
    return __builtin_amdgcn_wmma_f32_16x16x32_bf16(
        false, a.v, false, b.v, (short)0, c, false, false);
}

// ---------------------------------------------------------------------------
// fp32 [K,N] -> bf16 transposed [N,K]
// ---------------------------------------------------------------------------
__global__ void cvtT_f32_bf16(const float* __restrict__ src, u16* __restrict__ dst,
                              int K, int N) {
    int total = K * N;
    for (int i = blockIdx.x * blockDim.x + threadIdx.x; i < total;
         i += gridDim.x * blockDim.x) {
        int k = i / N, n = i - k * N;
        dst[(size_t)n * K + k] = f2bf(src[i]);
    }
}

// ---------------------------------------------------------------------------
// LayerNorm (fp32 in) -> bf16 out.  One block (256 thr) per row, D = 1024.
// ---------------------------------------------------------------------------
__global__ __launch_bounds__(256) void ln_bf16_kernel(
    const float* __restrict__ X, const float* __restrict__ g,
    const float* __restrict__ be, u16* __restrict__ out, int D) {
    int row = blockIdx.x;
    int tid = threadIdx.x;
    const float* x = X + (size_t)row * D;
    __shared__ float red[256];

    float s = 0.f;
    for (int i = tid; i < D; i += 256) s += x[i];
    red[tid] = s; __syncthreads();
    for (int off = 128; off > 0; off >>= 1) {
        if (tid < off) red[tid] += red[tid + off];
        __syncthreads();
    }
    float mean = red[0] / (float)D;
    __syncthreads();

    float v = 0.f;
    for (int i = tid; i < D; i += 256) { float d = x[i] - mean; v += d * d; }
    red[tid] = v; __syncthreads();
    for (int off = 128; off > 0; off >>= 1) {
        if (tid < off) red[tid] += red[tid + off];
        __syncthreads();
    }
    float rstd = rsqrtf(red[0] / (float)D + 1e-5f);

    u16* o = out + (size_t)row * D;
    for (int i = tid; i < D; i += 256)
        o[i] = f2bf((x[i] - mean) * rstd * g[i] + be[i]);
}

// ---------------------------------------------------------------------------
// bf16 WMMA GEMM: C[M,N] = A[M,K] @ W[K,N], with W given TRANSPOSED (Wt[N,K]).
// Block tile 128x128, 8 waves (4x2), wave tile 32x64 (2x4 WMMA 16x16x32).
// Epilogue: +bias, relu, +residual, fp32 out, bf16 out, bf16 transposed out
// (outbT laid out [B][N][Srows] for attention V).
// ---------------------------------------------------------------------------
#define GM 128
#define GN 128
#define GK 32

__global__ __launch_bounds__(256) void gemm_bf16_kernel(
    const u16* __restrict__ A, const u16* __restrict__ Wt,
    float* __restrict__ outf, u16* __restrict__ outb, u16* __restrict__ outbT,
    const float* __restrict__ bias, const float* __restrict__ resid,
    int M, int N, int K, int Srows, int do_relu) {
    __shared__ __align__(16) u16 Al[GM][GK];   // 8 KB
    __shared__ __align__(16) u16 Bl[GN][GK];   // 8 KB   Bl[n][k] = W[k][n] = Wt[n][k]

    int tid = threadIdx.x;
    int wid = tid >> 5, lane = tid & 31;
    int hl = lane >> 4, l16 = lane & 15;       // half-wave id, lane-in-half
    int waveM = wid >> 1, waveN = wid & 1;
    int m0 = blockIdx.y * GM, n0 = blockIdx.x * GN;

    v8f acc[2][4];
    for (int mi = 0; mi < 2; ++mi)
        for (int ni = 0; ni < 4; ++ni)
            for (int r = 0; r < 8; ++r) acc[mi][ni][r] = 0.f;

    // staging map: ids 0..511, row = id>>2, col = (id&3)*8 (both tiles 128x32)
    int r0 = tid >> 2,         c0 = (tid & 3) * 8;
    int r1 = (tid + 256) >> 2, c1 = ((tid + 256) & 3) * 8;

    for (int k0 = 0; k0 < K; k0 += GK) {
        __syncthreads();
        stage_b128(A  + (size_t)(m0 + r0) * K + k0 + c0, &Al[r0][c0]);
        stage_b128(A  + (size_t)(m0 + r1) * K + k0 + c1, &Al[r1][c1]);
        stage_b128(Wt + (size_t)(n0 + r0) * K + k0 + c0, &Bl[r0][c0]);
        stage_b128(Wt + (size_t)(n0 + r1) * K + k0 + c1, &Bl[r1][c1]);
        stage_wait();
        __syncthreads();

        BF16x16 af[2], bfr[4];
        // A frag: lane row = l16; k = hl*16 + i (contiguous, matches B)
        for (int mi = 0; mi < 2; ++mi) {
            const u16* p = &Al[waveM * 32 + mi * 16 + l16][hl * 16];
            af[mi].q[0] = *(const u32x4*)p;
            af[mi].q[1] = *(const u32x4*)(p + 8);
        }
        // B frag: lane col = l16; k = hl*16 + i (contiguous in Bl rows)
        for (int ni = 0; ni < 4; ++ni) {
            const u16* p = &Bl[waveN * 64 + ni * 16 + l16][hl * 16];
            bfr[ni].q[0] = *(const u32x4*)p;
            bfr[ni].q[1] = *(const u32x4*)(p + 8);
        }
        for (int mi = 0; mi < 2; ++mi)
            for (int ni = 0; ni < 4; ++ni)
                acc[mi][ni] = wmma_bf16(af[mi], bfr[ni], acc[mi][ni]);
    }

    // Epilogue: C layout — VGPR r, row = r + 8*hl, col = l16
    for (int mi = 0; mi < 2; ++mi)
        for (int ni = 0; ni < 4; ++ni) {
            int col = n0 + waveN * 64 + ni * 16 + l16;
            float bv = bias ? bias[col] : 0.f;
            for (int r = 0; r < 8; ++r) {
                int row = m0 + waveM * 32 + mi * 16 + r + hl * 8;
                float v = acc[mi][ni][r] + bv;
                if (do_relu) v = fmaxf(v, 0.f);
                size_t idx = (size_t)row * N + col;
                if (resid) v += resid[idx];
                if (outf) outf[idx] = v;
                if (outb) outb[idx] = f2bf(v);
                if (outbT) {
                    int bb = row / Srows, ss = row - bb * Srows;
                    outbT[((size_t)bb * N + col) * Srows + ss] = f2bf(v);
                }
            }
        }
}

// ---------------------------------------------------------------------------
// Flash attention. Q,K bf16 [B,S,D]; V bf16 TRANSPOSED [B,D,S].
// Block: one (b,h), 128 query rows; 8 waves x 16 rows. Key chunks of 32
// staged in LDS (all b128, no scatter). Online softmax, ctx written bf16.
// ---------------------------------------------------------------------------
__global__ __launch_bounds__(256) void attn_kernel(
    const u16* __restrict__ Qb, const u16* __restrict__ Kb,
    const u16* __restrict__ VbT, u16* __restrict__ ctxb,
    int S, int D, int H, int DK, float scale) {
    __shared__ __align__(16) u16 Kl[32][64];      // 4 KB  [key][d]
    __shared__ __align__(16) u16 Vt[64][32];      // 4 KB  [d][key]
    __shared__ __align__(16) u16 Pl[8][16][32];   // 8 KB  per-wave P relayout

    int bh = blockIdx.x;
    int b = bh / H, h = bh % H;
    int hoff = h * DK;
    int tid = threadIdx.x;
    int wid = tid >> 5, lane = tid & 31;
    int hl = lane >> 4, l16 = lane & 15;
    int q0 = blockIdx.y * 128 + wid * 16;
    size_t base  = (size_t)b * S * D + hoff;                 // Q/K/ctx rows
    size_t baseV = ((size_t)b * D + hoff) * S;               // V^T rows (d-major)

    // Q fragments: 16 rows x 64 d = 2 chunks of K=32
    BF16x16 qa[2];
    for (int c = 0; c < 2; ++c) {
        const u16* qp = Qb + base + (size_t)(q0 + l16) * D + c * 32 + hl * 16;
        qa[c].q[0] = *(const u32x4*)qp;
        qa[c].q[1] = *(const u32x4*)(qp + 8);
    }

    float mrow[8], lrow[8];
    v8f o[4];
    for (int r = 0; r < 8; ++r) { mrow[r] = -1e30f; lrow[r] = 0.f; }
    for (int j = 0; j < 4; ++j)
        for (int r = 0; r < 8; ++r) o[j][r] = 0.f;

    int kKey = tid >> 3, kD = (tid & 7) * 8;   // K staging: [key][d]
    int vD = tid >> 2,  vK = (tid & 3) * 8;    // V staging: [d][key] (contiguous in VbT)

    for (int kc = 0; kc < S; kc += 32) {
        __syncthreads();
        stage_b128(Kb + base + (size_t)(kc + kKey) * D + kD, &Kl[kKey][kD]);
        stage_b128(VbT + baseV + (size_t)vD * S + kc + vK,   &Vt[vD][vK]);
        stage_wait();
        if (kc + 32 < S) {
            __builtin_prefetch(Kb + base + (size_t)(kc + 32 + kKey) * D + kD, 0, 1);
            __builtin_prefetch(VbT + baseV + (size_t)vD * S + kc + 32 + vK, 0, 1);
        }
        __syncthreads();

        // S tile = Q(16x64) @ K^T(64x32): two 16x16 frags, 2 k-steps each
        v8f s0, s1;
        for (int r = 0; r < 8; ++r) { s0[r] = 0.f; s1[r] = 0.f; }
        for (int c = 0; c < 2; ++c) {
            BF16x16 bk0, bk1;
            const u16* p0 = &Kl[l16][c * 32 + hl * 16];
            bk0.q[0] = *(const u32x4*)p0; bk0.q[1] = *(const u32x4*)(p0 + 8);
            const u16* p1 = &Kl[16 + l16][c * 32 + hl * 16];
            bk1.q[0] = *(const u32x4*)p1; bk1.q[1] = *(const u32x4*)(p1 + 8);
            s0 = wmma_bf16(qa[c], bk0, s0);
            s1 = wmma_bf16(qa[c], bk1, s1);
        }

        // Online softmax per row (row = r + 8*hl; 32 cols live in 16-lane halves)
        for (int r = 0; r < 8; ++r) {
            float a0 = s0[r] * scale, a1 = s1[r] * scale;
            float rm = fmaxf(a0, a1);
            for (int mm = 8; mm >= 1; mm >>= 1) rm = fmaxf(rm, __shfl_xor(rm, mm, 32));
            float mnew = fmaxf(mrow[r], rm);
            float alpha = __expf(mrow[r] - mnew);
            float p0 = __expf(a0 - mnew), p1 = __expf(a1 - mnew);
            float psum = p0 + p1;
            for (int mm = 8; mm >= 1; mm >>= 1) psum += __shfl_xor(psum, mm, 32);
            lrow[r] = lrow[r] * alpha + psum;
            mrow[r] = mnew;
            for (int j = 0; j < 4; ++j) o[j][r] *= alpha;
            int rr = r + hl * 8;
            Pl[wid][rr][l16]      = f2bf(p0);
            Pl[wid][rr][16 + l16] = f2bf(p1);
        }

        // Re-read P as A-fragment (same-wave DS ops are in order)
        BF16x16 pa;
        {
            const u16* pp = &Pl[wid][l16][hl * 16];
            pa.q[0] = *(const u32x4*)pp;
            pa.q[1] = *(const u32x4*)(pp + 8);
        }
        // o += P(16x32) @ V(32x64): 4 n-tiles of 16
        for (int j = 0; j < 4; ++j) {
            BF16x16 bv;
            const u16* vp = &Vt[j * 16 + l16][hl * 16];
            bv.q[0] = *(const u32x4*)vp; bv.q[1] = *(const u32x4*)(vp + 8);
            o[j] = wmma_bf16(pa, bv, o[j]);
        }
    }

    // ctx = o / l, bf16 out, [B,S,D]
    for (int j = 0; j < 4; ++j)
        for (int r = 0; r < 8; ++r) {
            int row = q0 + r + hl * 8;
            ctxb[base + (size_t)row * D + j * 16 + l16] = f2bf(o[j][r] / lrow[r]);
        }
}

// ---------------------------------------------------------------------------
// Launch
// ---------------------------------------------------------------------------
extern "C" void kernel_launch(void* const* d_in, const int* in_sizes, int n_in,
                              void* d_out, int out_size, void* d_ws, size_t ws_size,
                              hipStream_t stream) {
    const float* X   = (const float*)d_in[0];
    const float* Wq  = (const float*)d_in[1];
    const float* Wk  = (const float*)d_in[2];
    const float* Wv  = (const float*)d_in[3];
    const float* Wo  = (const float*)d_in[4];
    const float* W1  = (const float*)d_in[5];
    const float* b1  = (const float*)d_in[6];
    const float* W2  = (const float*)d_in[7];
    const float* b2  = (const float*)d_in[8];
    const float* g1  = (const float*)d_in[9];
    const float* be1 = (const float*)d_in[10];
    const float* g2  = (const float*)d_in[11];
    const float* be2 = (const float*)d_in[12];

    const int B = 2, S = 2048, D = 1024, H = 16, DK = 64, FF = 4096;
    const int Mrows = B * S;

    char* ws = (char*)d_ws;
    size_t off = 0;
    auto alloc = [&](size_t bytes) -> void* {
        void* p = ws + off;
        off += (bytes + 255) & ~(size_t)255;
        return p;
    };
    u16* wqbT = (u16*)alloc((size_t)D * D * 2);   // [D(out)][D(in)]
    u16* wkbT = (u16*)alloc((size_t)D * D * 2);
    u16* wvbT = (u16*)alloc((size_t)D * D * 2);
    u16* wobT = (u16*)alloc((size_t)D * D * 2);
    u16* w1bT = (u16*)alloc((size_t)D * FF * 2);  // [FF][D]
    u16* w2bT = (u16*)alloc((size_t)FF * D * 2);  // [D][FF]
    u16* nxb  = (u16*)alloc((size_t)Mrows * D * 2);   // reused for both LNs
    u16* qb   = (u16*)alloc((size_t)Mrows * D * 2);
    u16* kb   = (u16*)alloc((size_t)Mrows * D * 2);
    u16* vbT  = (u16*)alloc((size_t)Mrows * D * 2);   // [B][D][S]
    u16* ctxb = (u16*)alloc((size_t)Mrows * D * 2);
    float* X1 = (float*)alloc((size_t)Mrows * D * 4);
    u16* hb   = qb;   // reuse Q/K/V/ctx region (32 MB) for FFN hidden after O-proj

    // 1) weights -> bf16, transposed to [N][K]
    cvtT_f32_bf16<<<2048, 256, 0, stream>>>(Wq, wqbT, D, D);
    cvtT_f32_bf16<<<2048, 256, 0, stream>>>(Wk, wkbT, D, D);
    cvtT_f32_bf16<<<2048, 256, 0, stream>>>(Wv, wvbT, D, D);
    cvtT_f32_bf16<<<2048, 256, 0, stream>>>(Wo, wobT, D, D);
    cvtT_f32_bf16<<<2048, 256, 0, stream>>>(W1, w1bT, D, FF);
    cvtT_f32_bf16<<<2048, 256, 0, stream>>>(W2, w2bT, FF, D);

    // 2) LN1
    ln_bf16_kernel<<<Mrows, 256, 0, stream>>>(X, g1, be1, nxb, D);

    // 3) Q/K/V projections (V written transposed for attention staging)
    dim3 gqkv(D / GN, Mrows / GM);
    gemm_bf16_kernel<<<gqkv, 256, 0, stream>>>(nxb, wqbT, nullptr, qb, nullptr,
                                               nullptr, nullptr, Mrows, D, D, S, 0);
    gemm_bf16_kernel<<<gqkv, 256, 0, stream>>>(nxb, wkbT, nullptr, kb, nullptr,
                                               nullptr, nullptr, Mrows, D, D, S, 0);
    gemm_bf16_kernel<<<gqkv, 256, 0, stream>>>(nxb, wvbT, nullptr, nullptr, vbT,
                                               nullptr, nullptr, Mrows, D, D, S, 0);

    // 4) flash attention
    dim3 ga(B * H, S / 128);
    attn_kernel<<<ga, 256, 0, stream>>>(qb, kb, vbT, ctxb, S, D, H, DK, 0.125f);

    // 5) O projection + residual -> X1
    gemm_bf16_kernel<<<gqkv, 256, 0, stream>>>(ctxb, wobT, X1, nullptr, nullptr,
                                               nullptr, X, Mrows, D, D, S, 0);

    // 6) LN2
    ln_bf16_kernel<<<Mrows, 256, 0, stream>>>(X1, g2, be2, nxb, D);

    // 7) FFN1: relu(nX2 @ W1 + b1) -> bf16 hidden
    dim3 gf1(FF / GN, Mrows / GM);
    gemm_bf16_kernel<<<gf1, 256, 0, stream>>>(nxb, w1bT, nullptr, hb, nullptr,
                                              b1, nullptr, Mrows, FF, D, S, 1);

    // 8) FFN2: X1 + h @ W2 + b2 -> d_out (fp32)
    dim3 gf2(D / GN, Mrows / GM);
    gemm_bf16_kernel<<<gf2, 256, 0, stream>>>(hb, w2bT, (float*)d_out, nullptr,
                                              nullptr, b2, X1, Mrows, D, FF, S, 0);
}
